// AttNMT_21139829030987
// MI455X (gfx1250) — matrix operand order repaired and checked by reading, not compiled, not run on hardware
//
#include <hip/hip_runtime.h>

typedef __bf16 bf16;
typedef __attribute__((ext_vector_type(16))) __bf16 v16bf;
typedef __attribute__((ext_vector_type(8)))  __bf16 v8bf;
typedef __attribute__((ext_vector_type(8)))  float  v8f;

#define DEVI __device__ __forceinline__

DEVI float sigf(float x) { return 1.0f / (1.0f + __expf(-x)); }

// ---------------------------------------------------------------------------
// bf16 WMMA GEMM:  C[M,N] (f32) = A[M,K](bf16) * W[N,K](bf16)^T + bias (+C if beta)
// One wave computes a 16(M) x 64(N) tile via 4 accumulators.
// Software-pipelined: next k-chunk's fragments are loaded before the current
// chunk's WMMAs execute, so s_wait_loadcnt only drains to the in-flight set
// and L2 latency overlaps with matrix ops.
// Fragment layouts used here (wave32; contiguous k order, not the ISA table order):
//   A 16x32: lane L -> M=L%16, K contiguous 16 @ (L<16?0:16)
//   B 32x16: lane L -> N=L%16, K contiguous 16 @ (L<16?0:16)
//   C 16x16: vgpr r -> M = r + (L<16?0:8), N = L%16
// ---------------------------------------------------------------------------
__global__ void __launch_bounds__(256)
k_gemm_bf16(const bf16* __restrict__ A, int lda,
            const bf16* __restrict__ W, int ldw,
            const float* __restrict__ bias,
            float* __restrict__ C, int ldc,
            int M, int N, int K, int beta)
{
    const int lane = threadIdx.x & 31;
    const int wave = blockIdx.x * (blockDim.x >> 5) + (threadIdx.x >> 5);
    const int nt64 = N >> 6;                 // N is always a multiple of 64 here
    const int mt   = wave / nt64;
    const int nt   = wave - mt * nt64;
    if (mt * 16 >= M) return;

    const int lm    = lane & 15;
    const int hi    = lane >> 4;             // 0 | 1
    const int koffA = hi * 16;
    const int koffB = hi * 16;

    const bf16* __restrict__ ap = A + (long)(mt * 16 + lm) * lda + koffA;
    const bf16* __restrict__ w0 = W + (long)(nt * 64 +  0 + lm) * ldw + koffB;
    const bf16* __restrict__ w1 = W + (long)(nt * 64 + 16 + lm) * ldw + koffB;
    const bf16* __restrict__ w2 = W + (long)(nt * 64 + 32 + lm) * ldw + koffB;
    const bf16* __restrict__ w3 = W + (long)(nt * 64 + 48 + lm) * ldw + koffB;

    v8f acc[4];
    if (beta) {
#pragma unroll
        for (int s = 0; s < 4; ++s)
#pragma unroll
            for (int r = 0; r < 8; ++r)
                acc[s][r] = C[(long)(mt * 16 + r + hi * 8) * ldc + nt * 64 + s * 16 + lm];
    } else {
#pragma unroll
        for (int s = 0; s < 4; ++s)
#pragma unroll
            for (int r = 0; r < 8; ++r)
                acc[s][r] = 0.0f;
    }

    // ---- prologue: load k-chunk 0 ----
    v8bf  alo = *(const v8bf*)(ap);
    v8bf  ahi = *(const v8bf*)(ap + 8);
    v16bf wf0 = *(const v16bf*)(w0);
    v16bf wf1 = *(const v16bf*)(w1);
    v16bf wf2 = *(const v16bf*)(w2);
    v16bf wf3 = *(const v16bf*)(w3);

    int k = 0;
    for (; k + 32 < K; k += 32) {
        // issue next-chunk loads first (they retire under the WMMAs below)
        v8bf  nalo = *(const v8bf*)(ap + k + 32);
        v8bf  nahi = *(const v8bf*)(ap + k + 40);
        v16bf nw0  = *(const v16bf*)(w0 + k + 32);
        v16bf nw1  = *(const v16bf*)(w1 + k + 32);
        v16bf nw2  = *(const v16bf*)(w2 + k + 32);
        v16bf nw3  = *(const v16bf*)(w3 + k + 32);
        __builtin_prefetch(w0 + k + 96, 0, 3);   // global_prefetch_b8, temporal (L2-resident weights)

        v16bf af = __builtin_shufflevector(alo, ahi,
                       0,1,2,3,4,5,6,7,8,9,10,11,12,13,14,15);
        acc[0] = __builtin_amdgcn_wmma_f32_16x16x32_bf16(false, af, false, wf0, (short)0, acc[0], false, false);
        acc[1] = __builtin_amdgcn_wmma_f32_16x16x32_bf16(false, af, false, wf1, (short)0, acc[1], false, false);
        acc[2] = __builtin_amdgcn_wmma_f32_16x16x32_bf16(false, af, false, wf2, (short)0, acc[2], false, false);
        acc[3] = __builtin_amdgcn_wmma_f32_16x16x32_bf16(false, af, false, wf3, (short)0, acc[3], false, false);

        alo = nalo; ahi = nahi;
        wf0 = nw0; wf1 = nw1; wf2 = nw2; wf3 = nw3;
    }
    // ---- epilogue chunk ----
    {
        v16bf af = __builtin_shufflevector(alo, ahi,
                       0,1,2,3,4,5,6,7,8,9,10,11,12,13,14,15);
        acc[0] = __builtin_amdgcn_wmma_f32_16x16x32_bf16(false, af, false, wf0, (short)0, acc[0], false, false);
        acc[1] = __builtin_amdgcn_wmma_f32_16x16x32_bf16(false, af, false, wf1, (short)0, acc[1], false, false);
        acc[2] = __builtin_amdgcn_wmma_f32_16x16x32_bf16(false, af, false, wf2, (short)0, acc[2], false, false);
        acc[3] = __builtin_amdgcn_wmma_f32_16x16x32_bf16(false, af, false, wf3, (short)0, acc[3], false, false);
    }

#pragma unroll
    for (int s = 0; s < 4; ++s) {
        const int n  = nt * 64 + s * 16 + lm;
        const float bv = bias ? bias[n] : 0.0f;
#pragma unroll
        for (int r = 0; r < 8; ++r)
            C[(long)(mt * 16 + r + hi * 8) * ldc + n] = acc[s][r] + bv;
    }
}

// ------------------------- elementwise / helper kernels ---------------------

__global__ void k_cvt_sub(const float* __restrict__ src, int src_ld, int col0,
                          bf16* __restrict__ dst, long n, int cols)
{
    long i = (long)blockIdx.x * blockDim.x + threadIdx.x;
    if (i >= n) return;
    long r = i / cols; int c = (int)(i - r * cols);
    dst[i] = (bf16)src[r * (long)src_ld + col0 + c];
}

__global__ void k_zero_f32(float* p, long n)
{ long i = (long)blockIdx.x * blockDim.x + threadIdx.x; if (i < n) p[i] = 0.0f; }

__global__ void k_zero_bf16(bf16* p, long n)
{ long i = (long)blockIdx.x * blockDim.x + threadIdx.x; if (i < n) p[i] = (bf16)0.0f; }

__global__ void k_gather_xemb(const float* __restrict__ emb, const int* __restrict__ idx,
                              const float* __restrict__ mask, bf16* __restrict__ dst,
                              long n, int cols)
{
    long i = (long)blockIdx.x * blockDim.x + threadIdx.x;
    if (i >= n) return;
    long r = i / cols; int e = (int)(i - r * cols);
    int tok = idx[r];
    dst[i] = (bf16)(emb[(long)tok * cols + e] * mask[r]);
}

__global__ void k_gather_yemb(const float* __restrict__ emb, const int* __restrict__ ydata,
                              int t, bf16* __restrict__ yemb,
                              bf16* __restrict__ decin, int ld_di,
                              bf16* __restrict__ readin_y, int ld_ri,
                              int B, int cols)
{
    int i = blockIdx.x * blockDim.x + threadIdx.x;
    if (i >= B * cols) return;
    int b = i / cols, e = i - b * cols;
    int tok = (t == 0) ? 0 : ydata[(t - 1) * B + b];
    bf16 v = (bf16)emb[(long)tok * cols + e];
    yemb[(long)b * cols + e]   = v;
    decin[(long)b * ld_di + e] = v;
    readin_y[(long)b * ld_ri + e] = v;
}

__global__ void k_lstm_gate(const float* __restrict__ zx, const float* __restrict__ zh,
                            float* __restrict__ c,
                            float* __restrict__ h_f32,
                            bf16* __restrict__ h_b, int ld_hb,
                            bf16* __restrict__ h_b2, int ld_hb2,
                            int B, int H)
{
    int i = blockIdx.x * blockDim.x + threadIdx.x;
    if (i >= B * H) return;
    int b = i / H, j = i - b * H;
    long z0 = (long)b * 4 * H;
    float ig = sigf(zx[z0 + j]         + zh[z0 + j]);
    float fg = sigf(zx[z0 + H + j]     + zh[z0 + H + j]);
    float gg = tanhf(zx[z0 + 2*H + j]  + zh[z0 + 2*H + j]);
    float og = sigf(zx[z0 + 3*H + j]   + zh[z0 + 3*H + j]);
    float cn = fg * c[i] + ig * gg;
    c[i] = cn;
    float h = og * tanhf(cn);
    if (h_f32) h_f32[i] = h;
    h_b[(long)b * ld_hb + j] = (bf16)h;
    if (h_b2) h_b2[(long)b * ld_hb2 + j] = (bf16)h;
}

__global__ void k_ctx_build(const float* __restrict__ of, const float* __restrict__ orv,
                            const float* __restrict__ xmask,
                            float* __restrict__ ctx, bf16* __restrict__ ctxb,
                            long n, int B, int E2, int Eh)
{
    long i = (long)blockIdx.x * blockDim.x + threadIdx.x;
    if (i >= n) return;
    long tb = i / E2; int e = (int)(i - tb * E2);
    float v = (e < Eh ? of[tb * Eh + e] : orv[tb * Eh + (e - Eh)]) * xmask[tb];
    ctx[i] = v; ctxb[i] = (bf16)v;
}

__global__ void k_ctx_mean(const float* __restrict__ ctx, const float* __restrict__ xmask,
                           float* __restrict__ mf, bf16* __restrict__ mb,
                           int Tx, int B, int C2)
{
    int i = blockIdx.x * blockDim.x + threadIdx.x;
    if (i >= B * C2) return;
    int b = i / C2, e = i - b * C2;
    float num = 0.0f, den = 0.0f;
    for (int t = 0; t < Tx; ++t) {
        num += ctx[((long)t * B + b) * C2 + e] * xmask[t * B + b];
        den += xmask[t * B + b];
    }
    float v = num / den;
    mf[i] = v; mb[i] = (bf16)v;
}

__global__ void k_tanh_act(const float* __restrict__ src, float* __restrict__ df,
                           bf16* __restrict__ db, long n)
{
    long i = (long)blockIdx.x * blockDim.x + threadIdx.x;
    if (i >= n) return;
    float v = tanhf(src[i]);
    if (df) df[i] = v;
    if (db) db[i] = (bf16)v;
}

// a2[tb] = dot(tanh(att_ctx[tb,:] + u[b,:]), att2_w) + att2_b  (one wave per row)
__global__ void k_att_score(const float* __restrict__ att_ctx, const float* __restrict__ u,
                            const float* __restrict__ w2, const float* __restrict__ b2,
                            float* __restrict__ a2, int TB, int B, int Ad)
{
    int wave = (blockIdx.x * blockDim.x + threadIdx.x) >> 5;
    int lane = threadIdx.x & 31;
    if (wave >= TB) return;
    int b = wave % B;
    const float* ac = att_ctx + (long)wave * Ad;
    const float* ub = u + (long)b * Ad;
    float s = 0.0f;
    for (int j = lane; j < Ad; j += 32)
        s += tanhf(ac[j] + ub[j]) * w2[j];
    for (int off = 16; off; off >>= 1) s += __shfl_xor(s, off, 32);
    if (lane == 0) a2[wave] = s + b2[0];
}

// softmax over Tx for each column b (one wave per b), masked
__global__ void k_softmax_col(const float* __restrict__ a2, const float* __restrict__ xmask,
                              float* __restrict__ alpha, int Tx, int B)
{
    int b = blockIdx.x, lane = threadIdx.x;
    float mx = -3.402823466e38f;
    for (int t = lane; t < Tx; t += 32) mx = fmaxf(mx, a2[t * B + b]);
    for (int off = 16; off; off >>= 1) mx = fmaxf(mx, __shfl_xor(mx, off, 32));
    float s = 0.0f;
    for (int t = lane; t < Tx; t += 32) {
        float e = __expf(a2[t * B + b] - mx) * xmask[t * B + b];
        alpha[t * B + b] = e;
        s += e;
    }
    for (int off = 16; off; off >>= 1) s += __shfl_xor(s, off, 32);
    float inv = 1.0f / (s + 1e-15f);
    for (int t = lane; t < Tx; t += 32) alpha[t * B + b] *= inv;
}

__global__ void k_ctx_weighted(const float* __restrict__ ctx, const float* __restrict__ alpha,
                               bf16* __restrict__ di_ctx, int ld_di,
                               bf16* __restrict__ ri_ctx, int ld_ri,
                               int Tx, int B, int C2)
{
    int b = blockIdx.x;
    for (int e = threadIdx.x; e < C2; e += blockDim.x) {
        float s = 0.0f;
        for (int t = 0; t < Tx; ++t)
            s += alpha[t * B + b] * ctx[((long)t * B + b) * C2 + e];
        bf16 v = (bf16)s;
        di_ctx[(long)b * ld_di + e] = v;
        ri_ctx[(long)b * ld_ri + e] = v;
    }
}

__global__ void k_maxout(const float* __restrict__ rw, bf16* __restrict__ r, int B, int Dw)
{
    int i = blockIdx.x * blockDim.x + threadIdx.x;
    if (i >= B * Dw) return;
    int b = i / Dw, j = i - b * Dw;
    r[i] = (bf16)fmaxf(rw[(long)b * 2 * Dw + 2 * j], rw[(long)b * 2 * Dw + 2 * j + 1]);
}

__global__ void k_nll(const float* __restrict__ logits, const int* __restrict__ ydata,
                      const float* __restrict__ ymask, int t,
                      float* __restrict__ losses, int B, int V)
{
    __shared__ float red[256];
    int b = blockIdx.x;
    const float* l = logits + (long)b * V;
    float mx = -3.402823466e38f;
    for (int j = threadIdx.x; j < V; j += 256) mx = fmaxf(mx, l[j]);
    red[threadIdx.x] = mx; __syncthreads();
    for (int s = 128; s; s >>= 1) { if (threadIdx.x < s) red[threadIdx.x] = fmaxf(red[threadIdx.x], red[threadIdx.x + s]); __syncthreads(); }
    mx = red[0]; __syncthreads();
    float sum = 0.0f;
    for (int j = threadIdx.x; j < V; j += 256) sum += expf(l[j] - mx);
    red[threadIdx.x] = sum; __syncthreads();
    for (int s = 128; s; s >>= 1) { if (threadIdx.x < s) red[threadIdx.x] += red[threadIdx.x + s]; __syncthreads(); }
    if (threadIdx.x == 0) {
        int tok = ydata[t * B + b];
        float logp = l[tok] - mx - logf(red[0]);
        losses[t * B + b] = -logp * ymask[t * B + b] * (1.0f / (float)B);
    }
}

__global__ void k_sum(const float* __restrict__ x, float* __restrict__ out, int n)
{
    __shared__ float red[256];
    float s = 0.0f;
    for (int i = threadIdx.x; i < n; i += 256) s += x[i];
    red[threadIdx.x] = s; __syncthreads();
    for (int st = 128; st; st >>= 1) { if (threadIdx.x < st) red[threadIdx.x] += red[threadIdx.x + st]; __syncthreads(); }
    if (threadIdx.x == 0) out[0] = red[0];
}

// ---------------------------------------------------------------------------

extern "C" void kernel_launch(void* const* d_in, const int* in_sizes, int n_in,
                              void* d_out, int out_size, void* d_ws, size_t ws_size,
                              hipStream_t stream)
{
    constexpr int Tx = 48, Ty = 48, Bn = 64;
    constexpr int DW = 512, E = 512, D = 1024, Ad = 512, VT = 32000;
    constexpr int E2 = 2 * E;             // 1024
    constexpr int DI = DW + E2;           // 1536 (dec_in)
    constexpr int RI = E2 + D + DW;       // 2560 (readin)

    const int*   x_data  = (const int*)  d_in[0];
    const float* x_mask  = (const float*)d_in[1];
    const int*   y_data  = (const int*)  d_in[2];
    const float* y_mask  = (const float*)d_in[3];
    const float* src_emb = (const float*)d_in[4];
    const float* dec_emb = (const float*)d_in[5];
    const float* Wih_f   = (const float*)d_in[6];
    const float* Whh_f   = (const float*)d_in[7];
    const float* b_f     = (const float*)d_in[8];
    const float* Wih_r   = (const float*)d_in[9];
    const float* Whh_r   = (const float*)d_in[10];
    const float* b_r     = (const float*)d_in[11];
    const float* h0_W    = (const float*)d_in[12];
    const float* h0_b    = (const float*)d_in[13];
    const float* c0_W    = (const float*)d_in[14];
    const float* c0_b    = (const float*)d_in[15];
    const float* att1_W  = (const float*)d_in[16];
    const float* att1_b  = (const float*)d_in[17];
    const float* att2_W  = (const float*)d_in[18];
    const float* att2_b  = (const float*)d_in[19];
    const float* cell_Wih= (const float*)d_in[20];
    const float* cell_Whh= (const float*)d_in[21];
    const float* cell_b  = (const float*)d_in[22];
    const float* read_W  = (const float*)d_in[23];
    const float* read_b  = (const float*)d_in[24];
    const float* out_W   = (const float*)d_in[25];
    const float* out_b   = (const float*)d_in[26];

    // --- bump allocator over workspace ---
    char* base = (char*)d_ws;
    size_t off = 0;
    auto carve = [&](size_t bytes) -> void* {
        void* p = base + off;
        off = (off + bytes + 255) & ~(size_t)255;
        return p;
    };
    auto cb = [&](long n) { return (bf16*)carve((size_t)n * sizeof(bf16)); };
    auto cf = [&](long n) { return (float*)carve((size_t)n * sizeof(float)); };

    bf16* xemb_b    = cb((long)Tx * Bn * DW);
    bf16* WihF_b    = cb((long)4*E * DW);
    bf16* WhhF_b    = cb((long)4*E * E);
    bf16* WihR_b    = cb((long)4*E * DW);
    bf16* WhhR_b    = cb((long)4*E * E);
    bf16* h0W_b     = cb((long)D * E2);
    bf16* c0W_b     = cb((long)D * E2);
    bf16* W1c_b     = cb((long)Ad * E2);
    bf16* W1y_b     = cb((long)Ad * DW);
    bf16* W1h_b     = cb((long)Ad * D);
    bf16* cWih_b    = cb((long)4*D * DI);
    bf16* cWhh_b    = cb((long)4*D * D);
    bf16* readW_b   = cb((long)2*DW * RI);
    bf16* outW_b    = cb((long)VT * DW);
    bf16* h_enc_b   = cb((long)Bn * E);
    bf16* ctx_b     = cb((long)Tx * Bn * E2);
    bf16* mean_b    = cb((long)Bn * E2);
    bf16* ht_b      = cb((long)Bn * D);
    bf16* yemb_b    = cb((long)Bn * DW);
    bf16* decin_b   = cb((long)Bn * DI);
    bf16* readin_b  = cb((long)Bn * RI);
    bf16* r_b       = cb((long)Bn * DW);

    float* zxF      = cf((long)Tx * Bn * 4*E);
    float* zxR      = cf((long)Tx * Bn * 4*E);
    float* zh_enc   = cf((long)Bn * 4*E);
    float* c_enc    = cf((long)Bn * E);
    float* out_f    = cf((long)Tx * Bn * E);
    float* out_r    = cf((long)Tx * Bn * E);
    float* ctx_f    = cf((long)Tx * Bn * E2);
    float* mean_f   = cf((long)Bn * E2);
    float* h0pre    = cf((long)Bn * D);
    float* c0pre    = cf((long)Bn * D);
    float* c_dec    = cf((long)Bn * D);
    float* attctx_f = cf((long)Tx * Bn * Ad);
    float* u_f      = cf((long)Bn * Ad);
    float* a2       = cf((long)Tx * Bn);
    float* alpha    = cf((long)Tx * Bn);
    float* zx_dec   = cf((long)Bn * 4*D);
    float* zh_dec   = cf((long)Bn * 4*D);
    float* rw_f     = cf((long)Bn * 2*DW);
    float* logits   = cf((long)Bn * VT);
    float* losses   = cf((long)Ty * Bn);
    (void)ws_size; (void)n_in; (void)in_sizes; (void)out_size;

    auto eg = [](long n) { return dim3((unsigned)((n + 255) / 256)); };
    auto gemm = [&](const bf16* A, int lda, const bf16* W, int ldw, const float* bias,
                    float* C, int ldc, int M, int N, int K, int beta) {
        int waves  = ((M + 15) / 16) * (N / 64);
        int blocks = (waves + 7) / 8;
        k_gemm_bf16<<<blocks, 256, 0, stream>>>(A, lda, W, ldw, bias, C, ldc, M, N, K, beta);
    };
    auto cvt = [&](const float* src, int sld, int c0, bf16* dst, long rows, int cols) {
        long n = rows * (long)cols;
        k_cvt_sub<<<eg(n), 256, 0, stream>>>(src, sld, c0, dst, n, cols);
    };

    // ---- weight conversion (fp32 -> bf16, once per call) ----
    cvt(Wih_f, DW, 0, WihF_b, 4*E, DW);
    cvt(Whh_f, E,  0, WhhF_b, 4*E, E);
    cvt(Wih_r, DW, 0, WihR_b, 4*E, DW);
    cvt(Whh_r, E,  0, WhhR_b, 4*E, E);
    cvt(h0_W,  E2, 0, h0W_b,  D, E2);
    cvt(c0_W,  E2, 0, c0W_b,  D, E2);
    cvt(att1_W, RI, 0,       W1c_b, Ad, E2);
    cvt(att1_W, RI, E2,      W1y_b, Ad, DW);
    cvt(att1_W, RI, E2 + DW, W1h_b, Ad, D);
    cvt(cell_Wih, DI, 0, cWih_b, 4*D, DI);
    cvt(cell_Whh, D,  0, cWhh_b, 4*D, D);
    cvt(read_W, RI, 0, readW_b, 2*DW, RI);
    cvt(out_W,  DW, 0, outW_b,  VT, DW);

    // ---- encoder ----
    long nxe = (long)Tx * Bn * DW;
    k_gather_xemb<<<eg(nxe), 256, 0, stream>>>(src_emb, x_data, x_mask, xemb_b, nxe, DW);

    // hoisted input-to-hidden GEMMs for all 48 steps (M = 3072)
    gemm(xemb_b, DW, WihF_b, DW, b_f, zxF, 4*E, Tx*Bn, 4*E, DW, 0);
    gemm(xemb_b, DW, WihR_b, DW, b_r, zxR, 4*E, Tx*Bn, 4*E, DW, 0);

    // forward direction
    k_zero_bf16<<<eg((long)Bn*E), 256, 0, stream>>>(h_enc_b, (long)Bn*E);
    k_zero_f32 <<<eg((long)Bn*E), 256, 0, stream>>>(c_enc,   (long)Bn*E);
    for (int t = 0; t < Tx; ++t) {
        gemm(h_enc_b, E, WhhF_b, E, nullptr, zh_enc, 4*E, Bn, 4*E, E, 0);
        k_lstm_gate<<<eg((long)Bn*E), 256, 0, stream>>>(
            zxF + (long)t*Bn*4*E, zh_enc, c_enc,
            out_f + (long)t*Bn*E, h_enc_b, E, nullptr, 0, Bn, E);
    }
    // reverse direction
    k_zero_bf16<<<eg((long)Bn*E), 256, 0, stream>>>(h_enc_b, (long)Bn*E);
    k_zero_f32 <<<eg((long)Bn*E), 256, 0, stream>>>(c_enc,   (long)Bn*E);
    for (int t = Tx - 1; t >= 0; --t) {
        gemm(h_enc_b, E, WhhR_b, E, nullptr, zh_enc, 4*E, Bn, 4*E, E, 0);
        k_lstm_gate<<<eg((long)Bn*E), 256, 0, stream>>>(
            zxR + (long)t*Bn*4*E, zh_enc, c_enc,
            out_r + (long)t*Bn*E, h_enc_b, E, nullptr, 0, Bn, E);
    }

    // ---- context / init state / attention precompute ----
    long nctx = (long)Tx * Bn * E2;
    k_ctx_build<<<eg(nctx), 256, 0, stream>>>(out_f, out_r, x_mask, ctx_f, ctx_b, nctx, Bn, E2, E);
    k_ctx_mean<<<eg((long)Bn*E2), 256, 0, stream>>>(ctx_f, x_mask, mean_f, mean_b, Tx, Bn, E2);

    gemm(mean_b, E2, h0W_b, E2, h0_b, h0pre, D, Bn, D, E2, 0);
    gemm(mean_b, E2, c0W_b, E2, c0_b, c0pre, D, Bn, D, E2, 0);
    k_tanh_act<<<eg((long)Bn*D), 256, 0, stream>>>(h0pre, nullptr, ht_b,  (long)Bn*D);
    k_tanh_act<<<eg((long)Bn*D), 256, 0, stream>>>(c0pre, c_dec,  nullptr,(long)Bn*D);

    gemm(ctx_b, E2, W1c_b, E2, att1_b, attctx_f, Ad, Tx*Bn, Ad, E2, 0);

    // ---- decoder ----
    for (int t = 0; t < Ty; ++t) {
        k_gather_yemb<<<eg((long)Bn*DW), 256, 0, stream>>>(
            dec_emb, y_data, t, yemb_b, decin_b, DI, readin_b + (E2 + D), RI, Bn, DW);

        // u = y_emb@W1y^T + ht@W1h^T
        gemm(yemb_b, DW, W1y_b, DW, nullptr, u_f, Ad, Bn, Ad, DW, 0);
        gemm(ht_b,   D,  W1h_b, D,  nullptr, u_f, Ad, Bn, Ad, D,  1);

        k_att_score<<<eg((long)Tx*Bn*32), 256, 0, stream>>>(
            attctx_f, u_f, att2_W, att2_b, a2, Tx*Bn, Bn, Ad);
        k_softmax_col<<<Bn, 32, 0, stream>>>(a2, x_mask, alpha, Tx, Bn);
        k_ctx_weighted<<<Bn, 256, 0, stream>>>(
            ctx_f, alpha, decin_b + DW, DI, readin_b, RI, Tx, Bn, E2);

        // LSTM cell
        gemm(decin_b, DI, cWih_b, DI, cell_b, zx_dec, 4*D, Bn, 4*D, DI, 0);
        gemm(ht_b,    D,  cWhh_b, D,  nullptr, zh_dec, 4*D, Bn, 4*D, D,  0);
        k_lstm_gate<<<eg((long)Bn*D), 256, 0, stream>>>(
            zx_dec, zh_dec, c_dec, nullptr, ht_b, D, readin_b + E2, RI, Bn, D);

        // readout + maxout
        gemm(readin_b, RI, readW_b, RI, read_b, rw_f, 2*DW, Bn, 2*DW, RI, 0);
        k_maxout<<<eg((long)Bn*DW), 256, 0, stream>>>(rw_f, r_b, Bn, DW);

        // vocab projection (dominant GEMM) + NLL
        gemm(r_b, DW, outW_b, DW, out_b, logits, VT, Bn, VT, DW, 0);
        k_nll<<<Bn, 256, 0, stream>>>(logits, y_data, y_mask, t, losses, Bn, VT);
    }

    k_sum<<<1, 256, 0, stream>>>(losses, (float*)d_out, Ty * Bn);
}
